// HeterogeneousODEDynamics_82308753261270
// MI455X (gfx1250) — matrix operand order repaired and checked by reading, not compiled, not run on hardware
//
#include <hip/hip_runtime.h>
#include <hip/hip_bf16.h>

// ---------------------------------------------------------------------------
// HeterogeneousODEDynamics on MI455X (gfx1250, wave32, WMMA).
// Strategy: bucket nodes by type (avoids the reference's 4x masked compute),
// run a fused 2-layer SiLU MLP per 128-row tile using v_wmma_f32_16x16x32_f16
// with f32 accumulation. Weights pre-converted to f16 K-major in workspace so
// B-fragments are single contiguous 32B loads (L2-resident, 192MB L2).
// Memory-bound: ~386MB HBM traffic -> ~16.6us floor @ 23.3 TB/s; f16 WMMA
// compute ~10us, safely under the memory floor.
// ---------------------------------------------------------------------------

#define N_NODES 262144
#define DIMD    128   // d
#define DIM2D   256   // 2d  (layer-1 K, also K-stride of transposed weights)
#define DIMH    256   // hidden
#define NTYPES  4
#define TILE_M  128
#define SXS     264   // padded LDS row stride in halves (avoids bank conflicts)

typedef _Float16 v16h __attribute__((ext_vector_type(16)));
typedef _Float16 v8h  __attribute__((ext_vector_type(8)));
typedef _Float16 v4h  __attribute__((ext_vector_type(4)));
typedef float    v8f  __attribute__((ext_vector_type(8)));

// ------------------------- bucketing kernels -------------------------------

__global__ void zero_meta(int* wsI) {
    if (threadIdx.x < 16) wsI[threadIdx.x] = 0;
}

__global__ void count_types(const int* __restrict__ types, int* __restrict__ wsI) {
    int i = blockIdx.x * blockDim.x + threadIdx.x;
    if (i < N_NODES) atomicAdd(&wsI[types[i]], 1);
}

// wsI[0..3]=counts, wsI[4..7]=padded offsets, wsI[8..11]=fill cursors
__global__ void make_offsets(int* wsI, int* bucket) {
    __shared__ int off[NTYPES + 1];
    if (threadIdx.x == 0) {
        int acc = 0;
        for (int t = 0; t < NTYPES; ++t) {
            wsI[4 + t] = acc;
            off[t] = acc;
            wsI[8 + t] = 0;  // cursor
            int padded = ((wsI[t] + TILE_M - 1) / TILE_M) * TILE_M;
            acc += padded;
        }
        off[NTYPES] = acc;
    }
    __syncthreads();
    // mark pad slots (beyond each type's real count) as invalid
    for (int t = 0; t < NTYPES; ++t) {
        int start = off[t] + wsI[t];
        int end   = off[t + 1];
        for (int j = start + (int)threadIdx.x; j < end; j += (int)blockDim.x)
            bucket[j] = -1;
    }
}

__global__ void fill_buckets(const int* __restrict__ types, int* wsI,
                             int* __restrict__ bucket) {
    int i = blockIdx.x * blockDim.x + threadIdx.x;
    if (i < N_NODES) {
        int t = types[i];
        int pos = atomicAdd(&wsI[8 + t], 1);
        bucket[wsI[4 + t] + pos] = i;
    }
}

// -------------------- weight pre-conversion (f32 -> f16, K-major) ----------
// W1:(T,2d,H) -> W1T:(T,H,2d)   W2:(T,H,d) -> W2T:(T,d,H)   (K contiguous)

__global__ void convert_weights(const float* __restrict__ W1,
                                const float* __restrict__ W2,
                                _Float16* __restrict__ W1T,
                                _Float16* __restrict__ W2T) {
    int gid = blockIdx.x * blockDim.x + threadIdx.x;
    const int nW1 = NTYPES * DIM2D * DIMH;  // 262144
    if (gid < nW1) {
        int t = gid / (DIM2D * DIMH);
        int r = gid % (DIM2D * DIMH);
        int hcol = r / DIM2D;  // output row  (n in [0,H))
        int k    = r % DIM2D;  // output col  (k in [0,2d))
        W1T[gid] = (_Float16)W1[t * DIM2D * DIMH + k * DIMH + hcol];
    } else {
        int g = gid - nW1;
        if (g < NTYPES * DIMH * DIMD) {
            int t = g / (DIMH * DIMD);
            int r = g % (DIMH * DIMD);
            int d = r / DIMH;  // output row (n in [0,d))
            int k = r % DIMH;  // output col (k in [0,H))
            W2T[g] = (_Float16)W2[t * DIMH * DIMD + k * DIMD + d];
        }
    }
}

// ------------------------- WMMA fragment loaders ---------------------------
// 16-bit A 16x32 layout (contiguous, matching B): lanes 0-15 -> M=lane, halves[0:16]=K0..15;
// lanes 16-31 -> M=lane-16, K16..31.
__device__ __forceinline__ v16h load_a_frag(const _Float16* sM, int rowBase,
                                            int kcol, int lane) {
    int m  = lane & 15;
    int k0 = (lane >> 4) << 4;  // 0 or 16
    const _Float16* p = sM + (rowBase + m) * SXS + kcol + k0;
    v8h lo = *(const v8h*)p;          // K = k0   .. k0+7
    v8h hi = *(const v8h*)(p + 8);    // K = k0+8 .. k0+15
    return __builtin_shufflevector(lo, hi, 0, 1, 2, 3, 4, 5, 6, 7,
                                           8, 9, 10, 11, 12, 13, 14, 15);
}

// 16-bit B 32x16: lanes 0-15 hold K=0..15 of column N=lane, lanes 16-31 hold
// K=16..31 of column N=lane-16 (contiguous k order, matching load_a_frag).
// Weights stored K-major (stride DIM2D halves) -> one contiguous 32B read.
__device__ __forceinline__ v16h load_b_frag(const _Float16* Wt, int ncol,
                                            int kcol, int lane) {
    int n  = ncol + (lane & 15);
    int k0 = kcol + ((lane >> 4) << 4);  // 0 or 16
    return *(const v16h*)(Wt + n * DIM2D + k0);
}

#define WMMA_F16(A, B, C) \
    __builtin_amdgcn_wmma_f32_16x16x32_f16(false, (A), false, (B), (short)0, (C), false, false)

// ------------------------------ fused MLP ----------------------------------

__global__ __launch_bounds__(256) void mlp_kernel(
    const float* __restrict__ hIn, const float* __restrict__ msgIn,
    const float* __restrict__ b1, const float* __restrict__ b2,
    const _Float16* __restrict__ W1T, const _Float16* __restrict__ W2T,
    const int* __restrict__ wsI, const int* __restrict__ bucket,
    float* __restrict__ out) {
    extern __shared__ char smem[];
    _Float16* sX = (_Float16*)smem;                 // 128 x 264 f16 (66KB)
    _Float16* sA = sX + TILE_M * SXS;               // 128 x 264 f16 (66KB)
    float* sB1 = (float*)(smem + 2 * TILE_M * SXS * sizeof(_Float16));
    float* sB2 = sB1 + DIMH;
    int* sIdx  = (int*)(sB2 + DIMD);

    const int tid  = threadIdx.x;
    const int lane = tid & 31;
    const int w    = tid >> 5;   // 8 waves
    const int mi   = w & 3;      // 4 M-groups of 32 rows
    const int ni   = w >> 2;     // 2 N-groups
    const int r0   = mi * 32;

    // ---- data-dependent tile assignment (uniform across block) ----
    int bid = blockIdx.x;
    int type = -1, base = 0, tb = 0;
    for (int t = 0; t < NTYPES; ++t) {
        int tiles = (wsI[t] + TILE_M - 1) >> 7;
        if (type < 0 && bid < tb + tiles) {
            type = t;
            base = wsI[4 + t] + (bid - tb) * TILE_M;
        }
        tb += tiles;
    }
    if (type < 0) return;  // whole block exits -> EXEC all-ones for WMMA below

    // ---- stage indices + biases ----
    if (tid < TILE_M) sIdx[tid] = bucket[base + tid];
    if (tid < DIMH) sB1[tid] = b1[type * DIMH + tid];
    if (tid < DIMD) sB2[tid] = b2[type * DIMD + tid];
    __syncthreads();

    // ---- gather x = [h || msg] rows, convert f32 -> f16 into LDS ----
    {
        int row  = tid >> 1;
        int half = tid & 1;  // 0: h cols 0..127, 1: msg cols 128..255
        int idx  = sIdx[row];
        _Float16* dst = sX + row * SXS + half * DIMD;
        if (idx >= 0) {
            const float4* src =
                (const float4*)((half ? msgIn : hIn) + (size_t)idx * DIMD);
#pragma unroll
            for (int j = 0; j < DIMD / 4; ++j) {
                float4 v = src[j];
                v4h hv = {(_Float16)v.x, (_Float16)v.y, (_Float16)v.z, (_Float16)v.w};
                *(v4h*)(dst + j * 4) = hv;
            }
        } else {
#pragma unroll
            for (int j = 0; j < DIMD / 4; ++j) { v4h z = {}; *(v4h*)(dst + j * 4) = z; }
        }
    }
    __syncthreads();

    // ---- layer 1: A1 = silu(X @ W1 + b1), X 128x256, W1 256x256 ----
    const _Float16* W1t = W1T + (size_t)type * DIM2D * DIMH;
#pragma unroll
    for (int nsub = 0; nsub < 4; ++nsub) {
        int ncol = ni * 128 + nsub * 32;
        v8f acc00 = {}, acc01 = {}, acc10 = {}, acc11 = {};
#pragma unroll
        for (int ks = 0; ks < 8; ++ks) {
            int kc = ks * 32;
            v16h a0 = load_a_frag(sX, r0, kc, lane);
            v16h a1 = load_a_frag(sX, r0 + 16, kc, lane);
            v16h bf0 = load_b_frag(W1t, ncol, kc, lane);
            v16h bf1 = load_b_frag(W1t, ncol + 16, kc, lane);
            acc00 = WMMA_F16(a0, bf0, acc00);
            acc01 = WMMA_F16(a0, bf1, acc01);
            acc10 = WMMA_F16(a1, bf0, acc10);
            acc11 = WMMA_F16(a1, bf1, acc11);
        }
        // bias + SiLU epilogue; C layout: vgpr r -> M=r+8*(lane>=16), N=lane&15
        int hiHalf = lane >> 4;
        int c0 = ncol + (lane & 15);
        float bias0 = sB1[c0], bias1 = sB1[c0 + 16];
#pragma unroll
        for (int r = 0; r < 8; ++r) {
            int row = r0 + r + 8 * hiHalf;
            float v0 = acc00[r] + bias0;
            float v1 = acc01[r] + bias1;
            sA[row * SXS + c0]      = (_Float16)(v0 / (1.f + __expf(-v0)));
            sA[row * SXS + c0 + 16] = (_Float16)(v1 / (1.f + __expf(-v1)));
            int row2 = row + 16;
            float v2 = acc10[r] + bias0;
            float v3 = acc11[r] + bias1;
            sA[row2 * SXS + c0]      = (_Float16)(v2 / (1.f + __expf(-v2)));
            sA[row2 * SXS + c0 + 16] = (_Float16)(v3 / (1.f + __expf(-v3)));
        }
    }
    __syncthreads();

    // ---- layer 2: out = A1 @ W2 + b2, A1 128x256, W2 256x128 ----
    const _Float16* W2t = W2T + (size_t)type * DIMH * DIMD;
#pragma unroll
    for (int nsub = 0; nsub < 2; ++nsub) {
        int ncol = ni * 64 + nsub * 32;
        v8f acc00 = {}, acc01 = {}, acc10 = {}, acc11 = {};
#pragma unroll
        for (int ks = 0; ks < 8; ++ks) {
            int kc = ks * 32;
            v16h a0 = load_a_frag(sA, r0, kc, lane);
            v16h a1 = load_a_frag(sA, r0 + 16, kc, lane);
            v16h bf0 = load_b_frag(W2t, ncol, kc, lane);
            v16h bf1 = load_b_frag(W2t, ncol + 16, kc, lane);
            acc00 = WMMA_F16(a0, bf0, acc00);
            acc01 = WMMA_F16(a0, bf1, acc01);
            acc10 = WMMA_F16(a1, bf0, acc10);
            acc11 = WMMA_F16(a1, bf1, acc11);
        }
        // bias + scatter back to global by node index (skip pad rows)
        int hiHalf = lane >> 4;
        int c0 = ncol + (lane & 15);
        float bias0 = sB2[c0], bias1 = sB2[c0 + 16];
#pragma unroll
        for (int r = 0; r < 8; ++r) {
            int rowL = r0 + r + 8 * hiHalf;
            int idxA = sIdx[rowL];
            if (idxA >= 0) {
                out[(size_t)idxA * DIMD + c0]      = acc00[r] + bias0;
                out[(size_t)idxA * DIMD + c0 + 16] = acc01[r] + bias1;
            }
            int rowL2 = rowL + 16;
            int idxB = sIdx[rowL2];
            if (idxB >= 0) {
                out[(size_t)idxB * DIMD + c0]      = acc10[r] + bias0;
                out[(size_t)idxB * DIMD + c0 + 16] = acc11[r] + bias1;
            }
        }
    }
}

// ------------------------------- launcher ----------------------------------

extern "C" void kernel_launch(void* const* d_in, const int* in_sizes, int n_in,
                              void* d_out, int out_size, void* d_ws, size_t ws_size,
                              hipStream_t stream) {
    (void)in_sizes; (void)n_in; (void)out_size; (void)ws_size;
    const float* h    = (const float*)d_in[0];
    const float* msg  = (const float*)d_in[1];
    const int*   tys  = (const int*)d_in[2];
    const float* W1   = (const float*)d_in[3];
    const float* b1   = (const float*)d_in[4];
    const float* W2   = (const float*)d_in[5];
    const float* b2   = (const float*)d_in[6];
    float* out = (float*)d_out;

    // workspace layout
    int* wsI    = (int*)d_ws;            // [0..15] meta (counts/offsets/cursors)
    int* bucket = wsI + 16;              // N + 4*TILE_M indices
    size_t wOff = 64 + (size_t)(N_NODES + NTYPES * TILE_M) * sizeof(int);
    wOff = (wOff + 255) & ~(size_t)255;
    _Float16* W1T = (_Float16*)((char*)d_ws + wOff);
    _Float16* W2T = W1T + NTYPES * DIM2D * DIMH;

    zero_meta<<<1, 32, 0, stream>>>(wsI);
    count_types<<<N_NODES / 256, 256, 0, stream>>>(tys, wsI);
    make_offsets<<<1, 256, 0, stream>>>(wsI, bucket);
    fill_buckets<<<N_NODES / 256, 256, 0, stream>>>(tys, wsI, bucket);
    convert_weights<<<(NTYPES * DIM2D * DIMH + NTYPES * DIMH * DIMD) / 256, 256,
                      0, stream>>>(W1, W2, W1T, W2T);

    size_t shmem = 2 * TILE_M * SXS * sizeof(_Float16) +
                   (DIMH + DIMD) * sizeof(float) + TILE_M * sizeof(int);  // 137216 B
    hipFuncSetAttribute((const void*)mlp_kernel,
                        hipFuncAttributeMaxDynamicSharedMemorySize, (int)shmem);
    mlp_kernel<<<N_NODES / TILE_M + NTYPES, 256, shmem, stream>>>(
        h, msg, b1, b2, W1T, W2T, wsI, bucket, out);
}
